// Critic_56934086475994
// MI455X (gfx1250) — matrix operand order repaired and checked by reading, compiled, not run on hardware
//
#include <hip/hip_runtime.h>

// ---------------------------------------------------------------------------
// MI455X (gfx1250) twin-critic GRU+MLP. All GEMMs via v_wmma_f32_16x16x32_f16.
// ---------------------------------------------------------------------------

typedef __attribute__((ext_vector_type(16))) _Float16 v16h;
typedef __attribute__((ext_vector_type(8)))  _Float16 h8;
typedef __attribute__((ext_vector_type(8)))  float    v8f;

union AF { v16h v; h8 h[2]; };

// A-matrix fragment (16x32 f16), CDNA5 layout:
//  lane<16 : M=lane,    halves[0..7]=K(k0+0..7),  halves[8..15]=K(k0+16..23)
//  lane>=16: M=lane-16, halves[0..7]=K(k0+8..15), halves[8..15]=K(k0+24..31)
static __device__ inline v16h load_a_frag(const _Float16* base, int ldh, int k0, int lane) {
    int m = lane & 15, g = lane >> 4;
    const _Float16* p = base + (long)m * ldh + k0 + g * 8;
    AF f;
    f.h[0] = *(const h8*)(p);
    f.h[1] = *(const h8*)(p + 16);
    return f.v;
}

// B-matrix fragment (32x16 f16) from row-major W[N][K]:
//  lane<16 : N=n0+lane,    halves[0..7]=K(k0+0..7),  halves[8..15]=K(k0+16..23)
//  lane>=16: N=n0+lane-16, halves[0..7]=K(k0+8..15), halves[8..15]=K(k0+24..31)
static __device__ inline v16h load_b_frag(const _Float16* base, int ldh, int n0, int k0, int lane) {
    int n = lane & 15, g = lane >> 4;
    const _Float16* p = base + (long)(n0 + n) * ldh + k0 + g * 8;
    AF f;
    f.h[0] = *(const h8*)(p);
    f.h[1] = *(const h8*)(p + 16);
    return f.v;
}

static __device__ inline v8f wmma16(v16h a, v16h b, v8f c) {
    return __builtin_amdgcn_wmma_f32_16x16x32_f16(false, a, false, b, (short)0, c, false, false);
}

static __device__ inline float sigmoidf_(float x) { return 1.f / (1.f + __expf(-x)); }

// ---------------------------------------------------------------------------
// fp32 -> f16 weight convert with K padding (zero-fill K in [Kin, Kout))
// ---------------------------------------------------------------------------
__global__ void cvt_pad(const float* __restrict__ src, _Float16* __restrict__ dst,
                        int N, int Kin, int Kout) {
    long i = (long)blockIdx.x * 256 + threadIdx.x;
    if (i >= (long)N * Kout) return;
    int n = (int)(i / Kout), k = (int)(i % Kout);
    dst[i] = (_Float16)(k < Kin ? src[(long)n * Kin + k] : 0.f);
}

// ---------------------------------------------------------------------------
// GRU over T steps, persistent h per 16-row batch tile.
//  Whh16: [768][256] f16, Wih16: [768][32] f16 (cols 15..31 zero)
//  Shared: hx = [h(0..255) | x_t(256..270) | zeros(271..287)] f16
//          S  : pre-activations f16 (cols 0..511 r/z combined,
//               512..767 gh_n, 768..1023 gi_n)
// ---------------------------------------------------------------------------
__global__ __launch_bounds__(256) void gru_kernel(
    const float* __restrict__ state, const int* __restrict__ lengths,
    const _Float16* __restrict__ Whh16, const _Float16* __restrict__ Wih16,
    const float* __restrict__ bih, const float* __restrict__ bhh,
    float* __restrict__ hout, int T, int D)
{
    __shared__ _Float16 hx[16][288];
    __shared__ _Float16 S[16][1024];
    __shared__ float    hf[16][256];

    const int tid  = threadIdx.x;
    const int lane = tid & 31;
    const int wave = tid >> 5;
    const int b0   = blockIdx.x * 16;

    for (int i = tid; i < 16 * 288; i += 256) (&hx[0][0])[i] = (_Float16)0.f;
    for (int i = tid; i < 16 * 256; i += 256) (&hf[0][0])[i] = 0.f;

    const int mrow = tid >> 4;          // 0..15 : batch row this thread updates
    const int c0   = (tid & 15) * 16;   // 16 hidden cols per thread
    const int len  = lengths[b0 + mrow];
    __syncthreads();

    for (int t = 0; t < T; ++t) {
        // stage x_t tile into hx[:, 256..270]
        if (tid < 16 * 15) {
            int m = tid / 15, d = tid % 15;
            hx[m][256 + d] = (_Float16)state[((long)(b0 + m) * T + t) * D + d];
        }
        __syncthreads();

        // --- WMMA phase: wave w owns N-tiles w*6 .. w*6+5 (48 tiles = 768 cols)
        for (int j = 0; j < 6; ++j) {
            const int n  = wave * 6 + j;      // wave-uniform
            const int nb = n * 16;
            const int col = nb + (lane & 15);
            const int r0  = (lane >> 4) * 8;
            if (n < 32) {                     // r,z gates: gi+gh combined
                v8f acc = {};
                for (int k = 0; k < 256; k += 32) {
                    v16h a = load_a_frag(&hx[0][0], 288, k, lane);
                    v16h b = load_b_frag(Whh16, 256, nb, k, lane);
                    acc = wmma16(a, b, acc);
                }
                {   // input-gate contribution (K=32, x slot of hx)
                    v16h a = load_a_frag(&hx[0][0], 288, 256, lane);
                    v16h b = load_b_frag(Wih16, 32, nb, 0, lane);
                    acc = wmma16(a, b, acc);
                }
                const float bsum = bih[col] + bhh[col];
                #pragma unroll
                for (int v = 0; v < 8; ++v)
                    S[r0 + v][col] = (_Float16)(acc[v] + bsum);
            } else {                          // n gate: gh_n and gi_n separate
                v8f acc = {};
                for (int k = 0; k < 256; k += 32) {
                    v16h a = load_a_frag(&hx[0][0], 288, k, lane);
                    v16h b = load_b_frag(Whh16, 256, nb, k, lane);
                    acc = wmma16(a, b, acc);
                }
                const float bh = bhh[col];
                #pragma unroll
                for (int v = 0; v < 8; ++v)
                    S[r0 + v][col] = (_Float16)(acc[v] + bh);
                v8f acc2 = {};
                {
                    v16h a = load_a_frag(&hx[0][0], 288, 256, lane);
                    v16h b = load_b_frag(Wih16, 32, nb, 0, lane);
                    acc2 = wmma16(a, b, acc2);
                }
                const float bi = bih[col];
                #pragma unroll
                for (int v = 0; v < 8; ++v)
                    S[r0 + v][col + 256] = (_Float16)(acc2[v] + bi);
            }
        }
        __syncthreads();

        // --- elementwise GRU update, ragged mask t < len
        if (t < len) {
            #pragma unroll 4
            for (int i = 0; i < 16; ++i) {
                const int c = c0 + i;
                const float ar  = (float)S[mrow][c];
                const float az  = (float)S[mrow][c + 256];
                const float ghn = (float)S[mrow][c + 512];
                const float gin = (float)S[mrow][c + 768];
                const float r  = sigmoidf_(ar);
                const float z  = sigmoidf_(az);
                const float nn = tanhf(gin + r * ghn);
                const float ho = hf[mrow][c];
                const float hn = (1.f - z) * nn + z * ho;
                hf[mrow][c] = hn;
                hx[mrow][c] = (_Float16)hn;
            }
        }
        __syncthreads();
    }

    #pragma unroll 4
    for (int i = 0; i < 16; ++i)
        hout[(long)(b0 + mrow) * 256 + c0 + i] = hf[mrow][c0 + i];
}

// ---------------------------------------------------------------------------
// Build X1 f16 [B][288] = [state[:,0,:](15) | action(1) | h(256) | pad(16)]
// ---------------------------------------------------------------------------
__global__ void build_x1(const float* __restrict__ state, const float* __restrict__ action,
                         const float* __restrict__ hsrc, _Float16* __restrict__ X1,
                         int T, int D) {
    long i = (long)blockIdx.x * 256 + threadIdx.x;
    if (i >= 2048L * 288) return;
    int b = (int)(i / 288), c = (int)(i % 288);
    float v;
    if (c < 15)       v = state[(long)b * T * D + c];     // t = 0
    else if (c == 15) v = action[b];
    else if (c < 272) v = hsrc[(long)b * 256 + (c - 16)];
    else              v = 0.f;
    X1[i] = (_Float16)v;
}

// ---------------------------------------------------------------------------
// GEMM + bias + ReLU: out[M][N] f16 = act(A[M][K] @ W[N][K]^T + bias)
// one 16x16 tile per wave, 8 waves -> 128 output cols per block
// ---------------------------------------------------------------------------
__global__ __launch_bounds__(256) void gemm_bias_act(
    const _Float16* __restrict__ A, const _Float16* __restrict__ W,
    const float* __restrict__ bias, _Float16* __restrict__ out,
    int M, int N, int K, int relu)
{
    const int lane = threadIdx.x & 31;
    const int wave = threadIdx.x >> 5;
    const int m0 = blockIdx.x * 16;
    const int n0 = blockIdx.y * 128 + wave * 16;
    if (n0 >= N || m0 >= M) return;          // wave-uniform
    const _Float16* At = A + (long)m0 * K;
    const _Float16* Wn = W + (long)n0 * K;
    v8f acc = {};
    for (int k = 0; k < K; k += 32) {
        __builtin_prefetch(Wn + (long)(lane & 15) * K + k + 128, 0, 1);
        v16h a = load_a_frag(At, K, k, lane);
        v16h b = load_b_frag(W, K, n0, k, lane);
        acc = wmma16(a, b, acc);
    }
    const int col = n0 + (lane & 15);
    const int r0  = (lane >> 4) * 8;
    const float bv = bias[col];
    #pragma unroll
    for (int v = 0; v < 8; ++v) {
        float x = acc[v] + bv;
        if (relu) x = fmaxf(x, 0.f);
        out[(long)(m0 + r0 + v) * N + col] = (_Float16)x;
    }
}

// ---------------------------------------------------------------------------
// q head: out[b] = act[b,:] . qw + qb   (256-dot, fp32)
// ---------------------------------------------------------------------------
__global__ void qhead(const _Float16* __restrict__ act, const float* __restrict__ qw,
                      const float* __restrict__ qb, float* __restrict__ out) {
    int b = blockIdx.x * 256 + threadIdx.x;
    if (b >= 2048) return;
    float s = qb[0];
    #pragma unroll 8
    for (int k = 0; k < 256; ++k) s += (float)act[(long)b * 256 + k] * qw[k];
    out[b] = s;
}

// ---------------------------------------------------------------------------
extern "C" void kernel_launch(void* const* d_in, const int* in_sizes, int n_in,
                              void* d_out, int out_size, void* d_ws, size_t ws_size,
                              hipStream_t stream) {
    const int B = 2048, T = 64, D = 15;
    const float* state  = (const float*)d_in[0];
    const float* action = (const float*)d_in[1];
    const int*   lengths = (const int*)d_in[2];

    char* ws = (char*)d_ws;
    size_t off = 0;
    auto alloc = [&](size_t bytes) -> void* {
        void* p = ws + off;
        off += (bytes + 255) & ~(size_t)255;
        return p;
    };

    dim3 blk(256);
    for (int g = 0; g < 2; ++g) {
        const int base = 3 + g * 14;
        const float* Wih  = (const float*)d_in[base + 0];
        const float* Whh  = (const float*)d_in[base + 1];
        const float* bih  = (const float*)d_in[base + 2];
        const float* bhh  = (const float*)d_in[base + 3];
        const float* w1 = (const float*)d_in[base + 4];
        const float* b1 = (const float*)d_in[base + 5];
        const float* w2 = (const float*)d_in[base + 6];
        const float* b2 = (const float*)d_in[base + 7];
        const float* w3 = (const float*)d_in[base + 8];
        const float* b3 = (const float*)d_in[base + 9];
        const float* w4 = (const float*)d_in[base + 10];
        const float* b4 = (const float*)d_in[base + 11];
        const float* qw = (const float*)d_in[base + 12];
        const float* qb = (const float*)d_in[base + 13];

        _Float16* Whh16 = (_Float16*)alloc(768L * 256 * 2);
        _Float16* Wih16 = (_Float16*)alloc(768L * 32 * 2);
        _Float16* W1p   = (_Float16*)alloc(1024L * 288 * 2);
        _Float16* W2p   = (_Float16*)alloc(1024L * 1024 * 2);
        _Float16* W3p   = (_Float16*)alloc(512L * 1024 * 2);
        _Float16* W4p   = (_Float16*)alloc(256L * 512 * 2);
        float*    hres  = (float*)   alloc((long)B * 256 * 4);
        _Float16* X1    = (_Float16*)alloc((long)B * 288 * 2);
        _Float16* bufA  = (_Float16*)alloc((long)B * 1024 * 2);
        _Float16* bufB  = (_Float16*)alloc((long)B * 1024 * 2);

        // weight conversion / padding
        cvt_pad<<<dim3((768 * 256 + 255) / 256), blk, 0, stream>>>(Whh, Whh16, 768, 256, 256);
        cvt_pad<<<dim3((768 * 32 + 255) / 256), blk, 0, stream>>>(Wih, Wih16, 768, 15, 32);
        cvt_pad<<<dim3((1024 * 288 + 255) / 256), blk, 0, stream>>>(w1, W1p, 1024, 272, 288);
        cvt_pad<<<dim3((1024 * 1024 + 255) / 256), blk, 0, stream>>>(w2, W2p, 1024, 1024, 1024);
        cvt_pad<<<dim3((512 * 1024 + 255) / 256), blk, 0, stream>>>(w3, W3p, 512, 1024, 1024);
        cvt_pad<<<dim3((256 * 512 + 255) / 256), blk, 0, stream>>>(w4, W4p, 256, 512, 512);

        // recurrent encoder
        gru_kernel<<<dim3(B / 16), blk, 0, stream>>>(state, lengths, Whh16, Wih16,
                                                     bih, bhh, hres, T, D);

        // MLP head
        build_x1<<<dim3((B * 288 + 255) / 256), blk, 0, stream>>>(state, action, hres, X1, T, D);
        gemm_bias_act<<<dim3(B / 16, 1024 / 128), blk, 0, stream>>>(X1,   W1p, b1, bufA, B, 1024, 288, 1);
        gemm_bias_act<<<dim3(B / 16, 1024 / 128), blk, 0, stream>>>(bufA, W2p, b2, bufB, B, 1024, 1024, 1);
        gemm_bias_act<<<dim3(B / 16,  512 / 128), blk, 0, stream>>>(bufB, W3p, b3, bufA, B, 512, 1024, 1);
        gemm_bias_act<<<dim3(B / 16,  256 / 128), blk, 0, stream>>>(bufA, W4p, b4, bufB, B, 256, 512, 1);
        qhead<<<dim3(B / 256), blk, 0, stream>>>(bufB, qw, qb, (float*)d_out + (long)g * B);
    }
}
